// Time_Features_Aware_MoE_9732395892812
// MI455X (gfx1250) — matrix operand order repaired and checked by reading, not compiled, not run on hardware
//
#include <hip/hip_runtime.h>
#include <hip/hip_bf16.h>
#include <math.h>

// ---------------------------------------------------------------------------
// Time_Features_Aware_MoE for MI455X (gfx1250, wave32, WMMA)
// One workgroup (256 threads = 8 waves) per batch element b.
// Pipeline staged entirely through LDS (~149 KB / WG -> 2 WGs per WGP).
// Gate1 (200x64 @ 64x128): each wave owns one fixed N-column (nt == wave),
// loads its two B fragments + bias ONCE into registers, then streams all 13
// M-tiles in pairs (independent WMMA chains fill the 7.12.1 hazard slots).
// Gate2 (200x128 @ 128x16): B fragments + bias hoisted out of the M loop.
// ---------------------------------------------------------------------------

typedef __attribute__((ext_vector_type(16))) _Float16 v16h;
typedef __attribute__((ext_vector_type(8)))  float    v8f;

#define Bc   4096
#define Lc   200
#define Hc   64
#define TDc  5
#define D1c  32
#define D2c  16
#define PERc 10
#define NLc  20
#define Ec   16
#define MTIL 13          // ceil(200/16)
#define MPAD 208         // MTIL*16

struct Smem {
    // ---- weights, resident for whole kernel (~46 KB) ----
    float w1[TDc * D1c];
    float b1[D1c]; float g1[D1c]; float be1[D1c];
    float w2[D1c * D2c];
    float b2[D2c]; float g2[D2c]; float be2[D2c];
    float convw[NLc * NLc * 9];
    float convb[NLc];
    float w3[D2c * Hc];
    float b3[Hc];
    float gb1[2 * Hc];
    float gb2[Ec];
    float experts[Ec * Hc];
    alignas(16) _Float16 gw1t[2 * Hc * Hc];     // transposed: [n=128][k=64] f16
    alignas(16) _Float16 gw2t[Ec * 2 * Hc];     // transposed: [n=16][k=128] f16
    // ---- staged activations ----
    float bufA[Lc * D2c];                        // h2, later logits
    float bufB[Lc * D2c];                        // conv output (already folded)
    alignas(16) _Float16 te[MPAD * Hc];          // time_emb f16 (WMMA A), padded rows zeroed
    union {
        alignas(16) _Float16 G[MPAD * 2 * Hc];   // gate hidden f16 (WMMA A for gate2)
        float mod[Lc * Hc];                      // route @ experts (after G is dead)
    } u;
};

// A fragment: 16x32 f16 tile at (row0, k0) of row-major src (leading dim ld).
// lane m=lane&15, hi=lane>>4; halves are one contiguous 16-half run:
//   regs u[0..7] = A[m][k0+16*hi .. +15]  (matches load_b_frag_t k order)
__device__ __forceinline__ v16h load_a_frag(const _Float16* src, int row0, int k0,
                                            int ld, int lane) {
    const int m  = lane & 15;
    const int hi = lane >> 4;
    const _Float16* row = src + (size_t)(row0 + m) * ld + k0;
    union { v16h v; uint4 q[2]; } r;
    r.q[0] = *(const uint4*)(row + 16 * hi);       // 16B, aligned
    r.q[1] = *(const uint4*)(row + 16 * hi + 8);   // 16B, aligned
    return r.v;
}

// B fragment from TRANSPOSED weights Bt[n][k] (k contiguous, leading dim ldk):
// lane n=lane&15, hi=lane>>4; halves h[0..15] = B[k0+16*hi .. +15][n0+n]
//  = Bt[n0+n][k0+16*hi .. +15]  -> one contiguous 32B run (two b128 loads).
__device__ __forceinline__ v16h load_b_frag_t(const _Float16* srcT, int k0, int n0,
                                              int ldk, int lane) {
    const int n  = lane & 15;
    const int hi = lane >> 4;
    const _Float16* p = srcT + (size_t)(n0 + n) * ldk + k0 + 16 * hi;
    union { v16h v; uint4 q[2]; } r;
    r.q[0] = *(const uint4*)(p);
    r.q[1] = *(const uint4*)(p + 8);
    return r.v;
}

// gate1 epilogue: bias (register scalar) + relu + f16 store into G (row-major).
__device__ __forceinline__ void store_g_tile(_Float16* G, const v8f& acc,
                                             int mt, int nt, float bias, int lane) {
    const int n  = lane & 15;
    const int hi = lane >> 4;
    _Float16* p = G + (size_t)(mt * 16 + 8 * hi) * (2 * Hc) + nt * 16 + n;
#pragma unroll
    for (int r = 0; r < 8; ++r) {
        float v = acc[r] + bias;
        p[r * (2 * Hc)] = (_Float16)(v > 0.f ? v : 0.f);
    }
}

__global__ __launch_bounds__(256)
void tfa_moe_kernel(const float* __restrict__ seq, const float* __restrict__ tf,
                    const float* __restrict__ w1, const float* __restrict__ b1,
                    const float* __restrict__ g1, const float* __restrict__ be1,
                    const float* __restrict__ w2, const float* __restrict__ b2,
                    const float* __restrict__ g2, const float* __restrict__ be2,
                    const float* __restrict__ convw, const float* __restrict__ convb,
                    const float* __restrict__ w3, const float* __restrict__ b3,
                    const float* __restrict__ gw1, const float* __restrict__ gb1,
                    const float* __restrict__ gw2, const float* __restrict__ gb2,
                    const float* __restrict__ experts,
                    float* __restrict__ out) {
    __shared__ Smem sm;
    const int tid  = threadIdx.x;
    const int lane = tid & 31;
    const int wave = tid >> 5;
    const int b    = blockIdx.x;

    const size_t base = (size_t)b * Lc * Hc;
    const float* seqb = seq + base;

    // Prefetch this WG's seq_emb slice toward cache while the gate pipe runs
    // (lowers to global_prefetch_b8 on gfx1250).
    for (int i = tid; i < Lc; i += 256)
        __builtin_prefetch(seqb + (size_t)i * Hc, 0, 1);

    // ---- load weights into LDS (L2-resident across all 4096 WGs) ----
    for (int i = tid; i < TDc * D1c; i += 256) sm.w1[i] = w1[i];
    for (int i = tid; i < D1c; i += 256) { sm.b1[i] = b1[i]; sm.g1[i] = g1[i]; sm.be1[i] = be1[i]; }
    for (int i = tid; i < D1c * D2c; i += 256) sm.w2[i] = w2[i];
    for (int i = tid; i < D2c; i += 256) { sm.b2[i] = b2[i]; sm.g2[i] = g2[i]; sm.be2[i] = be2[i]; sm.gb2[i] = gb2[i]; }
    for (int i = tid; i < NLc * NLc * 9; i += 256) sm.convw[i] = convw[i];
    for (int i = tid; i < NLc; i += 256) sm.convb[i] = convb[i];
    for (int i = tid; i < D2c * Hc; i += 256) { sm.w3[i] = w3[i]; sm.experts[i] = experts[i]; }
    for (int i = tid; i < Hc; i += 256) sm.b3[i] = b3[i];
    for (int i = tid; i < 2 * Hc; i += 256) sm.gb1[i] = gb1[i];
    // transposed f16 copies of the gate weights (B operands): Bt[n][k] = B[k][n]
    for (int i = tid; i < 2 * Hc * Hc; i += 256) {          // gw1t: n=i>>6, k=i&63
        const int n = i >> 6, k = i & (Hc - 1);
        sm.gw1t[i] = (_Float16)gw1[k * (2 * Hc) + n];
    }
    for (int i = tid; i < Ec * 2 * Hc; i += 256) {          // gw2t: n=i>>7, k=i&127
        const int n = i >> 7, k = i & (2 * Hc - 1);
        sm.gw2t[i] = (_Float16)gw2[k * Ec + n];
    }
    __syncthreads();

    // ---- Phase A: per-token MLP 5->32 (LN,ReLU) -> 16 (LN,ReLU) -> bufA ----
    if (tid < Lc) {
        const float* tfp = tf + ((size_t)b * Lc + tid) * TDc;
        float tfl[TDc];
#pragma unroll
        for (int k = 0; k < TDc; ++k) tfl[k] = tfp[k];

        float h1[D1c];
        float mean = 0.f;
#pragma unroll
        for (int c = 0; c < D1c; ++c) {
            float z = sm.b1[c];
#pragma unroll
            for (int k = 0; k < TDc; ++k) z += tfl[k] * sm.w1[k * D1c + c];
            h1[c] = z; mean += z;
        }
        mean *= (1.f / D1c);
        float var = 0.f;
#pragma unroll
        for (int c = 0; c < D1c; ++c) { float d = h1[c] - mean; var += d * d; }
        float rstd = rsqrtf(var * (1.f / D1c) + 1e-5f);
#pragma unroll
        for (int c = 0; c < D1c; ++c) {
            float v = (h1[c] - mean) * rstd * sm.g1[c] + sm.be1[c];
            h1[c] = v > 0.f ? v : 0.f;
        }

        float h2[D2c];
        mean = 0.f;
#pragma unroll
        for (int d = 0; d < D2c; ++d) {
            float z = sm.b2[d];
#pragma unroll
            for (int c = 0; c < D1c; ++c) z += h1[c] * sm.w2[c * D2c + d];
            h2[d] = z; mean += z;
        }
        mean *= (1.f / D2c);
        var = 0.f;
#pragma unroll
        for (int d = 0; d < D2c; ++d) { float dd = h2[d] - mean; var += dd * dd; }
        rstd = rsqrtf(var * (1.f / D2c) + 1e-5f);
#pragma unroll
        for (int d = 0; d < D2c; ++d) {
            float v = (h2[d] - mean) * rstd * sm.g2[d] + sm.be2[d];
            sm.bufA[tid * D2c + d] = v > 0.f ? v : 0.f;
        }
    }
    __syncthreads();

    // ---- Phase B: 3x3 SAME conv over [NL=20][F=16][P=10]; fold-back = flat copy ----
    // input[nl][f][p] = h2[l = nl*10+p][f]  (from bufA); output flat index o -> bufB[o]
    for (int o = tid; o < NLc * D2c * PERc; o += 256) {
        const int oc  = o / (D2c * PERc);
        const int rem = o - oc * (D2c * PERc);
        const int f   = rem / PERc;
        const int p   = rem - f * PERc;
        float acc = sm.convb[oc];
        for (int ic = 0; ic < NLc; ++ic) {
            const float* wk = &sm.convw[(oc * NLc + ic) * 9];
#pragma unroll
            for (int df = -1; df <= 1; ++df) {
                const int ff = f + df;
                if (ff < 0 || ff >= D2c) continue;
#pragma unroll
                for (int dp = -1; dp <= 1; ++dp) {
                    const int pp = p + dp;
                    if (pp < 0 || pp >= PERc) continue;
                    acc += sm.bufA[(ic * PERc + pp) * D2c + ff] * wk[(df + 1) * 3 + (dp + 1)];
                }
            }
        }
        sm.bufB[o] = acc;
    }
    __syncthreads();

    // ---- Phase C: time_emb = bufB(200x16) @ w3(16x64) + b3 -> f16 te (pad rows 0) ----
    for (int l = tid; l < MPAD; l += 256) {
        union { _Float16 h[Hc]; uint4 q[Hc / 8]; } row;
        if (l < Lc) {
            float xr[D2c];
#pragma unroll
            for (int d = 0; d < D2c; ++d) xr[d] = sm.bufB[l * D2c + d];
#pragma unroll
            for (int c = 0; c < Hc; ++c) {
                float acc = sm.b3[c];
#pragma unroll
                for (int d = 0; d < D2c; ++d) acc += xr[d] * sm.w3[d * Hc + c];
                row.h[c] = (_Float16)acc;
            }
        } else {
#pragma unroll
            for (int c = 0; c < Hc; ++c) row.h[c] = (_Float16)0.f;
        }
        uint4* dq = (uint4*)&sm.te[l * Hc];
#pragma unroll
        for (int j = 0; j < Hc / 8; ++j) dq[j] = row.q[j];   // 8x ds_store_b128
    }
    __syncthreads();

    // ---- Phase D: gate1 = relu(te @ gw1 + gb1) via WMMA.
    //      nt == wave: B fragments + bias live in registers for all 13 M-tiles;
    //      M-tiles processed in pairs to fill WMMA hazard slots. ----
    {
        const int nt    = wave;                               // 8 waves == 8 N-tiles
        const float bias = sm.gb1[nt * 16 + (lane & 15)];     // lane-const all phase
        const v16h bk0  = load_b_frag_t(sm.gw1t,  0, nt * 16, Hc, lane);
        const v16h bk1  = load_b_frag_t(sm.gw1t, 32, nt * 16, Hc, lane);
#pragma unroll 1
        for (int mt = 0; mt < 12; mt += 2) {                  // 6 pairs
            v16h a00 = load_a_frag(sm.te, mt * 16,        0, Hc, lane);
            v16h a01 = load_a_frag(sm.te, mt * 16,       32, Hc, lane);
            v16h a10 = load_a_frag(sm.te, (mt + 1) * 16,  0, Hc, lane);
            v16h a11 = load_a_frag(sm.te, (mt + 1) * 16, 32, Hc, lane);
            v8f acc0 = {}, acc1 = {};
            acc0 = __builtin_amdgcn_wmma_f32_16x16x32_f16(false, a00, false, bk0,
                                                          (short)0, acc0, false, false);
            acc1 = __builtin_amdgcn_wmma_f32_16x16x32_f16(false, a10, false, bk0,
                                                          (short)0, acc1, false, false);
            acc0 = __builtin_amdgcn_wmma_f32_16x16x32_f16(false, a01, false, bk1,
                                                          (short)0, acc0, false, false);
            acc1 = __builtin_amdgcn_wmma_f32_16x16x32_f16(false, a11, false, bk1,
                                                          (short)0, acc1, false, false);
            store_g_tile(sm.u.G, acc0, mt,     nt, bias, lane);
            store_g_tile(sm.u.G, acc1, mt + 1, nt, bias, lane);
        }
        {                                                     // remainder: mt = 12
            v16h a0 = load_a_frag(sm.te, 12 * 16,  0, Hc, lane);
            v16h a1 = load_a_frag(sm.te, 12 * 16, 32, Hc, lane);
            v8f acc = {};
            acc = __builtin_amdgcn_wmma_f32_16x16x32_f16(false, a0, false, bk0,
                                                         (short)0, acc, false, false);
            acc = __builtin_amdgcn_wmma_f32_16x16x32_f16(false, a1, false, bk1,
                                                         (short)0, acc, false, false);
            store_g_tile(sm.u.G, acc, 12, nt, bias, lane);
        }
    }
    __syncthreads();

    // ---- Phase E: logits = G(208x128) @ gw2(128x16) + gb2 via WMMA; 13 M-tiles.
    //      B fragments (4 K-steps) + bias hoisted out of the M loop. ----
    {
        const int n  = lane & 15;
        const int hi = lane >> 4;
        const float bias2 = sm.gb2[n];
        v16h bb[4];
#pragma unroll
        for (int ks = 0; ks < 4; ++ks)
            bb[ks] = load_b_frag_t(sm.gw2t, ks * 32, 0, 2 * Hc, lane);
        for (int t = wave; t < MTIL; t += 8) {
            v8f acc = {};
#pragma unroll
            for (int ks = 0; ks < 4; ++ks) {
                v16h af = load_a_frag(sm.u.G, t * 16, ks * 32, 2 * Hc, lane);
                acc = __builtin_amdgcn_wmma_f32_16x16x32_f16(false, af, false, bb[ks],
                                                             (short)0, acc, false, false);
            }
#pragma unroll
            for (int r = 0; r < 8; ++r) {
                const int m = t * 16 + r + 8 * hi;
                if (m < Lc) sm.bufA[m * Ec + n] = acc[r] + bias2;
            }
        }
    }
    __syncthreads();

    // ---- Phase F: softmax + top-2 routing -> mod = route @ experts ----
    if (tid < Lc) {
        float lg[Ec];
#pragma unroll
        for (int i = 0; i < Ec; ++i) lg[i] = sm.bufA[tid * Ec + i];
        float mx = lg[0];
#pragma unroll
        for (int i = 1; i < Ec; ++i) mx = lg[i] > mx ? lg[i] : mx;
        float den = 0.f;
#pragma unroll
        for (int i = 0; i < Ec; ++i) den += expf(lg[i] - mx);
        int i1 = 0; float v1 = lg[0];
#pragma unroll
        for (int i = 1; i < Ec; ++i) if (lg[i] > v1) { v1 = lg[i]; i1 = i; }
        int i2 = -1; float v2 = -3.4e38f;
#pragma unroll
        for (int i = 0; i < Ec; ++i) if (i != i1 && lg[i] > v2) { v2 = lg[i]; i2 = i; }
        const float inv = 1.f / den;
        const float p1 = expf(v1 - mx) * inv;
        const float p2 = expf(v2 - mx) * inv;
        const float* e1 = &sm.experts[i1 * Hc];
        const float* e2 = &sm.experts[i2 * Hc];
#pragma unroll
        for (int c = 0; c < Hc; ++c)
            sm.u.mod[tid * Hc + c] = p1 * e1[c] + p2 * e2[c];
    }
    __syncthreads();

    // ---- Final: out = seq_emb * mod  (coalesced stream) ----
    for (int i = tid; i < Lc * Hc; i += 256)
        out[base + i] = seqb[i] * sm.u.mod[i];
}

extern "C" void kernel_launch(void* const* d_in, const int* in_sizes, int n_in,
                              void* d_out, int out_size, void* d_ws, size_t ws_size,
                              hipStream_t stream) {
    (void)in_sizes; (void)n_in; (void)out_size; (void)d_ws; (void)ws_size;
    const float* seq     = (const float*)d_in[0];
    const float* tf      = (const float*)d_in[1];
    const float* w1      = (const float*)d_in[2];
    const float* b1      = (const float*)d_in[3];
    const float* g1      = (const float*)d_in[4];
    const float* be1     = (const float*)d_in[5];
    const float* w2      = (const float*)d_in[6];
    const float* b2      = (const float*)d_in[7];
    const float* g2      = (const float*)d_in[8];
    const float* be2     = (const float*)d_in[9];
    const float* convw   = (const float*)d_in[10];
    const float* convb   = (const float*)d_in[11];
    const float* w3      = (const float*)d_in[12];
    const float* b3      = (const float*)d_in[13];
    const float* gw1     = (const float*)d_in[14];
    const float* gb1     = (const float*)d_in[15];
    const float* gw2     = (const float*)d_in[16];
    const float* gb2     = (const float*)d_in[17];
    const float* experts = (const float*)d_in[18];
    float* out = (float*)d_out;

    tfa_moe_kernel<<<dim3(Bc), dim3(256), 0, stream>>>(
        seq, tf, w1, b1, g1, be1, w2, b2, g2, be2,
        convw, convb, w3, b3, gw1, gb1, gw2, gb2, experts, out);
}
